// SpatioTemporalAttention_34462817583419
// MI455X (gfx1250) — matrix operand order repaired and checked by reading, compiled, not run on hardware
//
#include <hip/hip_runtime.h>
#include <hip/hip_bf16.h>
#include <math.h>

// Problem constants (match reference)
#define B_    2
#define L_    12
#define N_    256
#define C_    512
#define H_    8
#define D_    64
#define T_    (L_ * N_)      // 3072
#define COND_ 64
#define SIGMA2_ 64.0f        // 8^2
#define SCALE_  0.125f       // 64^-0.5
#define EPS_    1e-5f

typedef _Float16 f16;
typedef __attribute__((ext_vector_type(16))) _Float16 v16h;
typedef __attribute__((ext_vector_type(8)))  _Float16 v8h;
typedef __attribute__((ext_vector_type(8)))  float    v8f;
typedef __attribute__((ext_vector_type(4)))  unsigned int v4u;
typedef __attribute__((ext_vector_type(8)))  int v8i;
typedef __attribute__((ext_vector_type(4)))  int v4i;

// ---------------------------------------------------------------------------
// WMMA fragment helpers (CDNA5 V_WMMA_F32_16X16X32_F16, wave32)
//
// A (16x32 f16) layout per ISA 7.12.2: lane L holds row M=L%16.
//   lanes 0-15 : halfs[0..7]=K0..7,  halfs[8..15]=K16..23
//   lanes 16-31: halfs[0..7]=K8..15, halfs[8..15]=K24..31
// B (32x16 f16): lane L holds column N=L%16.
//   lanes 0-15 : halfs[0..7]=K0..7,  halfs[8..15]=K16..23
//   lanes 16-31: halfs[0..7]=K8..15, halfs[8..15]=K24..31
//   (loaded from a K-contiguous "B-transposed" buffer: row = N, 32 K's per row)
// C/D (16x16 f32): lanes 0-15: vgpr r -> (M=r, N=lane); lanes 16-31: (M=8+r).
// ---------------------------------------------------------------------------

__device__ __forceinline__ int lane_id() { return (int)(threadIdx.x & 31u); }

__device__ __forceinline__ v16h load_a_frag(const f16* base, int ld) {
  int lane = lane_id();
  const f16* p = base + (lane & 15) * ld + (lane >> 4) * 8;
  v8h lo = *(const v8h*)(p);
  v8h hi = *(const v8h*)(p + 16);
  v16h r;
#pragma unroll
  for (int i = 0; i < 8; ++i) { r[i] = lo[i]; r[i + 8] = hi[i]; }
  return r;
}

__device__ __forceinline__ v16h load_bt_frag(const f16* bt, int ld) {
  // bt is [N][K] with K contiguous (row stride = ld halves, multiple of 8)
  int lane = lane_id();
  const f16* p = bt + (lane & 15) * ld + (lane >> 4) * 8;
  v8h lo = *(const v8h*)(p);
  v8h hi = *(const v8h*)(p + 16);
  v16h r;
#pragma unroll
  for (int i = 0; i < 8; ++i) { r[i] = lo[i]; r[i + 8] = hi[i]; }
  return r;
}

__device__ __forceinline__ v8f wmma_f16(v16h a, v16h b, v8f c) {
  return __builtin_amdgcn_wmma_f32_16x16x32_f16(
      /*neg_a=*/false, a, /*neg_b=*/false, b,
      /*c_mod=*/(short)0, c, /*reuse_a=*/false, /*reuse_b=*/false);
}

// Async global->LDS b128 copy (one 16-byte chunk per lane). Tracked by ASYNCcnt.
__device__ __forceinline__ void async_copy_b128(unsigned int lds_addr,
                                                unsigned long long gaddr) {
  asm volatile("global_load_async_to_lds_b128 %0, %1, off"
               :: "v"(lds_addr), "v"(gaddr) : "memory");
}
__device__ __forceinline__ void wait_asynccnt0() {
  asm volatile("s_wait_asynccnt 0x0" ::: "memory");
}

// ---------------------------------------------------------------------------
// Kernel 1: ss = cond @ adaln_w + adaln_b   -> [B, 2C]
// ---------------------------------------------------------------------------
__global__ void k_adaln_ss(const float* __restrict__ cond,
                           const float* __restrict__ w,
                           const float* __restrict__ bias,
                           float* __restrict__ ss) {
  __shared__ float sc[COND_];
  int b = blockIdx.x;
  if (threadIdx.x < COND_) sc[threadIdx.x] = cond[b * COND_ + threadIdx.x];
  __syncthreads();
  for (int j = threadIdx.x; j < 2 * C_; j += blockDim.x) {
    float acc = bias[j];
#pragma unroll 8
    for (int c = 0; c < COND_; ++c) acc += sc[c] * w[c * 2 * C_ + j];
    ss[b * 2 * C_ + j] = acc;
  }
}

// ---------------------------------------------------------------------------
// Kernel 2: AdaLN: non-affine LN over C, modulate, emit f16 x  [B*T, C]
// ---------------------------------------------------------------------------
__global__ __launch_bounds__(256) void k_adaln_norm(const float* __restrict__ x,
                                                    const float* __restrict__ ss,
                                                    f16* __restrict__ xo) {
  int row = blockIdx.x;                 // b*T + t
  int b = row / T_;
  const float* xr = x + (size_t)row * C_;
  int tid = threadIdx.x;
  float x0 = xr[tid], x1 = xr[tid + 256];
  __shared__ float rs[256], rq[256];
  rs[tid] = x0 + x1;
  rq[tid] = x0 * x0 + x1 * x1;
  __syncthreads();
  for (int s = 128; s > 0; s >>= 1) {
    if (tid < s) { rs[tid] += rs[tid + s]; rq[tid] += rq[tid + s]; }
    __syncthreads();
  }
  float mu   = rs[0] * (1.0f / C_);
  float var  = rq[0] * (1.0f / C_) - mu * mu;
  float rinv = rsqrtf(var + EPS_);
  const float* sc = ss + b * 2 * C_;
  float y0 = (x0 - mu) * rinv * (1.0f + sc[tid])       + sc[C_ + tid];
  float y1 = (x1 - mu) * rinv * (1.0f + sc[tid + 256]) + sc[C_ + tid + 256];
  xo[(size_t)row * C_ + tid]       = (f16)y0;
  xo[(size_t)row * C_ + tid + 256] = (f16)y1;
}

// ---------------------------------------------------------------------------
// Kernel 3: pack wq|wk|wv -> f16 [C, 3C], bq|bk|bv -> f32 [3C], wo -> f16
// ---------------------------------------------------------------------------
__global__ void k_pack(const float* __restrict__ wq, const float* __restrict__ bq,
                       const float* __restrict__ wk, const float* __restrict__ bk,
                       const float* __restrict__ wv, const float* __restrict__ bv,
                       const float* __restrict__ wo,
                       f16* __restrict__ wqkv, float* __restrict__ bqkv,
                       f16* __restrict__ wof) {
  int idx = blockIdx.x * blockDim.x + threadIdx.x;
  int stride = gridDim.x * blockDim.x;
  for (int i = idx; i < C_ * 3 * C_; i += stride) {
    int c = i / (3 * C_), j = i % (3 * C_);
    float v = (j < C_)      ? wq[c * C_ + j]
            : (j < 2 * C_)  ? wk[c * C_ + (j - C_)]
                            : wv[c * C_ + (j - 2 * C_)];
    wqkv[i] = (f16)v;
  }
  for (int i = idx; i < C_ * C_; i += stride) wof[i] = (f16)wo[i];
  for (int i = idx; i < 3 * C_; i += stride)
    bqkv[i] = (i < C_) ? bq[i] : (i < 2 * C_) ? bk[i - C_] : bv[i - 2 * C_];
}

// ---------------------------------------------------------------------------
// Kernel 4: generic WMMA GEMM  Out[M,N] = A[M,K](f16) @ Bw[K,N](f16) + bias
// block = 256 threads (8 waves); block tile 64x64; wave tile 16x32; K step 32
// A tile staged with async global->LDS copies (ASYNCcnt path).
// ---------------------------------------------------------------------------
__global__ __launch_bounds__(256) void k_gemm(const f16* __restrict__ A,
                                              const f16* __restrict__ Bw,
                                              const float* __restrict__ bias,
                                              float* __restrict__ Out,
                                              int M, int N, int K) {
  __shared__ __align__(16) f16 sA[64 * 32];    // [m][k]
  __shared__ __align__(16) f16 sBt[64 * 32];   // [n][k]  (transposed stage)
  int tid  = threadIdx.x;
  int wid  = tid >> 5;
  int lane = tid & 31;
  int m0 = blockIdx.x * 64, n0 = blockIdx.y * 64;
  int mw = (wid & 3) * 16;          // wave row tile
  int nw = (wid >> 2) * 32;         // wave col tile (2 x 16)
  v8f acc0 = {}; v8f acc1 = {};

  int ra = tid >> 2, c8 = (tid & 3) * 8;
  unsigned int sA_dst =
      (unsigned int)(uintptr_t)(void*)&sA[ra * 32 + c8];   // LDS byte offset

  for (int k0 = 0; k0 < K; k0 += 32) {
    __syncthreads();
    // stage A: 64x32 halves via async DMA into LDS (one b128 per lane)
    async_copy_b128(sA_dst,
        (unsigned long long)(uintptr_t)&A[(size_t)(m0 + ra) * K + k0 + c8]);
    {   // stage B transposed: thread gathers 8 K's of one column
      int n = tid & 63, kk = (tid >> 6) * 8;
      v8h tmp;
#pragma unroll
      for (int j = 0; j < 8; ++j)
        tmp[j] = Bw[(size_t)(k0 + kk + j) * N + n0 + n];
      *(v8h*)&sBt[n * 32 + kk] = tmp;
    }
    wait_asynccnt0();
    __syncthreads();
    v16h a  = load_a_frag(&sA[mw * 32], 32);
    v16h b0 = load_bt_frag(&sBt[(nw + 0)  * 32], 32);
    v16h b1 = load_bt_frag(&sBt[(nw + 16) * 32], 32);
    acc0 = wmma_f16(a, b0, acc0);
    acc1 = wmma_f16(a, b1, acc1);
  }

  int hi = lane >> 4, nn = lane & 15;
  int mb = m0 + mw + hi * 8;
  int col0 = n0 + nw + nn;
  float bb0 = bias[col0], bb1 = bias[col0 + 16];
#pragma unroll
  for (int r = 0; r < 8; ++r) {
    Out[(size_t)(mb + r) * N + col0]      = acc0[r] + bb0;
    Out[(size_t)(mb + r) * N + col0 + 16] = acc1[r] + bb1;
  }
}

// ---------------------------------------------------------------------------
// Kernel 5: 2D RoPE + head reshape.
//   qkv f32 [B*T, 3C] -> q/k/v f16 [B,H,T,D]; also f32 k,v into d_out slices.
// block = 256 threads = 8 heads x 32 rotation pairs; one (b,t) per block
// ---------------------------------------------------------------------------
__global__ __launch_bounds__(256) void k_rope(const float* __restrict__ qkv,
                                              const int* __restrict__ frame_idx,
                                              const int* __restrict__ seq_idx,
                                              f16* __restrict__ qh, f16* __restrict__ kh,
                                              f16* __restrict__ vh,
                                              float* __restrict__ kout,
                                              float* __restrict__ vout) {
  int row = blockIdx.x;                 // b*T + t
  int b = row / T_, t = row % T_;
  int l = t / N_, n = t % N_;
  int tid = threadIdx.x;
  int h = tid >> 5, p = tid & 31;
  int half = p >> 4, j = p & 15;        // half 0: seq rotation, half 1: frame
  int d0 = half * 32 + j, d1 = d0 + 16;
  float pos  = (half == 0) ? (float)seq_idx[b * N_ + n] : (float)frame_idx[l];
  float invf = __powf(10000.0f, -(float)j / 16.0f);
  float th = pos * invf;
  float cs = __cosf(th), sn = __sinf(th);

  const float* base = qkv + (size_t)row * (3 * C_);
  int hd = h * D_;
  float q0 = base[hd + d0],          q1 = base[hd + d1];
  float k0 = base[C_ + hd + d0],     k1 = base[C_ + hd + d1];
  float v0 = base[2 * C_ + hd + d0], v1 = base[2 * C_ + hd + d1];
  float qr0 = q0 * cs - q1 * sn, qr1 = q1 * cs + q0 * sn;
  float kr0 = k0 * cs - k1 * sn, kr1 = k1 * cs + k0 * sn;

  size_t o = ((size_t)(b * H_ + h) * T_ + t) * D_;
  qh[o + d0] = (f16)qr0;  qh[o + d1] = (f16)qr1;
  kh[o + d0] = (f16)kr0;  kh[o + d1] = (f16)kr1;
  vh[o + d0] = (f16)v0;   vh[o + d1] = (f16)v1;
  kout[o + d0] = kr0;     kout[o + d1] = kr1;
  vout[o + d0] = v0;      vout[o + d1] = v1;
}

// ---------------------------------------------------------------------------
// Kernel 6: flash attention with block-causal frame mask + Gaussian bias.
// grid = B*H*(L*N/64); block = 128 threads (4 waves x 16 query rows).
// K tile staged by the Tensor Data Mover (TENSORcnt path) where available;
// V staged transposed by VALU; S and PV via v_wmma_f32_16x16x32_f16.
// ---------------------------------------------------------------------------
#define QTPF_ (N_ / 64)                 // 4 query tiles per frame

__global__ __launch_bounds__(128) void k_attn(const f16* __restrict__ qh,
                                              const f16* __restrict__ kh,
                                              const f16* __restrict__ vh,
                                              const float* __restrict__ ca,
                                              const float* __restrict__ mask,
                                              f16* __restrict__ out) {
  int bid = blockIdx.x;
  int b   = bid / (H_ * L_ * QTPF_);
  int rem = bid % (H_ * L_ * QTPF_);
  int h   = rem / (L_ * QTPF_);
  int qt  = rem % (L_ * QTPF_);
  int l   = qt / QTPF_;
  int t0  = l * N_ + (qt % QTPF_) * 64;
  int kvlen = (l + 1) * N_;             // block-causal over frames

  __shared__ __align__(16) f16 sK[32 * 64];      // [kv][c]
  __shared__ __align__(16) f16 sVt[64 * 32];     // [d][kv] (transposed)
  __shared__ float sCk[32 * 3];
  __shared__ float sMk[32];
  __shared__ __align__(16) f16 sP[4][16 * 32];   // per-wave P staging

  int tid  = threadIdx.x;
  int wid  = tid >> 5;
  int lane = tid & 31;
  int hi   = lane >> 4, nn = lane & 15;

  size_t headoff = (size_t)(b * H_ + h) * T_;

  // Q fragments for this wave's 16 rows (kept in VGPRs across the kv loop)
  const f16* qbase = qh + (headoff + t0 + wid * 16) * D_;
  v16h aq0 = load_a_frag(qbase + 0, D_);
  v16h aq1 = load_a_frag(qbase + 32, D_);

  // CA positions of the 8 query rows this lane's accumulators cover
  float caq[8][3];
  {
    int mrow = t0 + wid * 16 + hi * 8;
#pragma unroll
    for (int r = 0; r < 8; ++r) {
      const float* cp = ca + ((size_t)b * T_ + mrow + r) * 3;
      caq[r][0] = cp[0]; caq[r][1] = cp[1]; caq[r][2] = cp[2];
    }
  }

  v8f o0 = {}, o1 = {}, o2 = {}, o3 = {};
  float mrow[8], lrow[8];
#pragma unroll
  for (int r = 0; r < 8; ++r) { mrow[r] = -1e30f; lrow[r] = 0.0f; }

  unsigned int sK_lds = (unsigned int)(uintptr_t)(void*)&sK[0];

  for (int kv0 = 0; kv0 < kvlen; kv0 += 32) {
    __syncthreads();
#if __has_builtin(__builtin_amdgcn_tensor_load_to_lds) && \
    __has_builtin(__builtin_amdgcn_s_wait_tensorcnt)
    // --- K tile via Tensor Data Mover: 2D tile 64 x 32 of 2-byte elements ---
    if (wid == 0) {
      unsigned long long gaddr =
          (unsigned long long)(uintptr_t)(kh + (headoff + kv0) * D_);
      v4u g0;
      g0[0] = 1u;                                   // count=1, user mode
      g0[1] = sK_lds;                               // lds_addr (bytes)
      g0[2] = (unsigned int)gaddr;                  // global_addr[31:0]
      g0[3] = (unsigned int)((gaddr >> 32) & 0x01FFFFFFu) | 0x80000000u; // type=2
      v8i g1;
      g1[0] = 0x00010000;                           // data_size = 2 bytes
      g1[1] = (int)(64u << 16);                     // tensor_dim0 = 64 (lo16)
      g1[2] = 0;                                    // td0 hi, td1 lo
      g1[3] = (int)((64u << 16) | 0x10u);           // tile_dim0=64, td1=1<<20
      g1[4] = 32;                                   // tile_dim1 = 32
      g1[5] = 64;                                   // tensor_dim0_stride = 64
      g1[6] = 0;
      g1[7] = 0;
      v4i gz4 = {0, 0, 0, 0};
      v8i gz8 = {0, 0, 0, 0, 0, 0, 0, 0};
      __builtin_amdgcn_tensor_load_to_lds(g0, g1, gz4, gz4, gz8, 0);
      __builtin_amdgcn_s_wait_tensorcnt(0);
    }
#else
    {   // fallback: manual K staging
      int r = tid >> 2, c16 = (tid & 3) * 16;
      const f16* kp = kh + (headoff + kv0 + r) * D_ + c16;
      *(v8h*)&sK[r * 64 + c16]     = *(const v8h*)(kp);
      *(v8h*)&sK[r * 64 + c16 + 8] = *(const v8h*)(kp + 8);
    }
#endif
    {   // stage V transposed [64][32] + ca/mask of kv tile
      int r = tid >> 2, c16 = (tid & 3) * 16;
      const f16* vp = vh + (headoff + kv0 + r) * D_ + c16;
#pragma unroll
      for (int j = 0; j < 16; ++j) sVt[(c16 + j) * 32 + r] = vp[j];
      if (tid < 32) {
        const float* cp = ca + ((size_t)b * T_ + kv0 + tid) * 3;
        sCk[tid * 3 + 0] = cp[0];
        sCk[tid * 3 + 1] = cp[1];
        sCk[tid * 3 + 2] = cp[2];
        sMk[tid] = mask[(size_t)b * T_ + kv0 + tid];
      }
    }
    __syncthreads();

    // S = Q K^T : two 16x16 tiles (kv cols 0-15 and 16-31), K-dim 64 = 2x32
    v8f s0 = {}, s1 = {};
    {
      v16h b00 = load_bt_frag(&sK[0 * 64 + 0],  64);
      v16h b01 = load_bt_frag(&sK[0 * 64 + 32], 64);
      s0 = wmma_f16(aq0, b00, s0);
      s0 = wmma_f16(aq1, b01, s0);
      v16h b10 = load_bt_frag(&sK[16 * 64 + 0],  64);
      v16h b11 = load_bt_frag(&sK[16 * 64 + 32], 64);
      s1 = wmma_f16(aq0, b10, s1);
      s1 = wmma_f16(aq1, b11, s1);
    }

    // bias: scale, -dist^2/sigma^2, pad mask
    float ck0x = sCk[nn * 3 + 0], ck0y = sCk[nn * 3 + 1], ck0z = sCk[nn * 3 + 2];
    float ck1x = sCk[(16 + nn) * 3 + 0], ck1y = sCk[(16 + nn) * 3 + 1],
          ck1z = sCk[(16 + nn) * 3 + 2];
    float pm0 = (sMk[nn]      == 0.0f) ? -1e30f : 0.0f;
    float pm1 = (sMk[16 + nn] == 0.0f) ? -1e30f : 0.0f;
#pragma unroll
    for (int r = 0; r < 8; ++r) {
      float dx = caq[r][0] - ck0x, dy = caq[r][1] - ck0y, dz = caq[r][2] - ck0z;
      float d2 = dx * dx + dy * dy + dz * dz;
      s0[r] = s0[r] * SCALE_ - d2 * (1.0f / SIGMA2_) + pm0;
      dx = caq[r][0] - ck1x; dy = caq[r][1] - ck1y; dz = caq[r][2] - ck1z;
      d2 = dx * dx + dy * dy + dz * dz;
      s1[r] = s1[r] * SCALE_ - d2 * (1.0f / SIGMA2_) + pm1;
    }

    // online softmax (row reductions across the 16 lanes holding each row)
    float al[8];
#pragma unroll
    for (int r = 0; r < 8; ++r) {
      float v = fmaxf(s0[r], s1[r]);
      v = fmaxf(v, __shfl_xor(v, 1));
      v = fmaxf(v, __shfl_xor(v, 2));
      v = fmaxf(v, __shfl_xor(v, 4));
      v = fmaxf(v, __shfl_xor(v, 8));
      float mn = fmaxf(mrow[r], v);
      al[r] = __expf(mrow[r] - mn);
      mrow[r] = mn;
      float p0 = __expf(s0[r] - mn);
      float p1 = __expf(s1[r] - mn);
      s0[r] = p0; s1[r] = p1;
      float rs = p0 + p1;
      rs += __shfl_xor(rs, 1);
      rs += __shfl_xor(rs, 2);
      rs += __shfl_xor(rs, 4);
      rs += __shfl_xor(rs, 8);
      lrow[r] = lrow[r] * al[r] + rs;
    }
#pragma unroll
    for (int r = 0; r < 8; ++r) {
      o0[r] *= al[r]; o1[r] *= al[r]; o2[r] *= al[r]; o3[r] *= al[r];
    }

    // re-layout P (C-layout -> A-layout) through per-wave LDS
    f16* P = &sP[wid][0];
    asm volatile("" ::: "memory");
#pragma unroll
    for (int r = 0; r < 8; ++r) {
      P[(hi * 8 + r) * 32 + nn]      = (f16)s0[r];
      P[(hi * 8 + r) * 32 + 16 + nn] = (f16)s1[r];
    }
    asm volatile("s_wait_dscnt 0x0" ::: "memory");
    v16h ap  = load_a_frag(P, 32);
    v16h bv0 = load_bt_frag(&sVt[0  * 32], 32);
    v16h bv1 = load_bt_frag(&sVt[16 * 32], 32);
    v16h bv2 = load_bt_frag(&sVt[32 * 32], 32);
    v16h bv3 = load_bt_frag(&sVt[48 * 32], 32);
    o0 = wmma_f16(ap, bv0, o0);
    o1 = wmma_f16(ap, bv1, o1);
    o2 = wmma_f16(ap, bv2, o2);
    o3 = wmma_f16(ap, bv3, o3);
  }

  // normalize and emit [B*T, H*D] f16 for the output projection
  int mbase = t0 + wid * 16 + hi * 8;
#pragma unroll
  for (int r = 0; r < 8; ++r) {
    float inv = 1.0f / lrow[r];
    size_t rowo = ((size_t)b * T_ + mbase + r) * C_ + h * D_;
    out[rowo + 0 * 16 + nn] = (f16)(o0[r] * inv);
    out[rowo + 1 * 16 + nn] = (f16)(o1[r] * inv);
    out[rowo + 2 * 16 + nn] = (f16)(o2[r] * inv);
    out[rowo + 3 * 16 + nn] = (f16)(o3[r] * inv);
  }
}

// ---------------------------------------------------------------------------
// Launch
// ---------------------------------------------------------------------------
extern "C" void kernel_launch(void* const* d_in, const int* in_sizes, int n_in,
                              void* d_out, int out_size, void* d_ws, size_t ws_size,
                              hipStream_t stream) {
  (void)in_sizes; (void)n_in; (void)out_size; (void)ws_size;
  const float* s_frames  = (const float*)d_in[0];
  const int*   frame_idx = (const int*)d_in[1];
  const int*   seq_idx   = (const int*)d_in[2];
  const float* mask      = (const float*)d_in[3];
  const float* cond      = (const float*)d_in[4];
  const float* ca_pos    = (const float*)d_in[5];
  const float* adaln_w   = (const float*)d_in[6];
  const float* adaln_b   = (const float*)d_in[7];
  const float* wq = (const float*)d_in[8];
  const float* bq = (const float*)d_in[9];
  const float* wk = (const float*)d_in[10];
  const float* bk = (const float*)d_in[11];
  const float* wv = (const float*)d_in[12];
  const float* bv = (const float*)d_in[13];
  const float* wo = (const float*)d_in[14];
  const float* bo = (const float*)d_in[15];

  char* ws = (char*)d_ws;
  size_t off = 0;
  auto alloc = [&](size_t bytes) -> void* {
    void* p = ws + off;
    off = (off + bytes + 255) & ~(size_t)255;
    return p;
  };
  float* ss    = (float*)alloc((size_t)B_ * 2 * C_ * 4);
  f16*   xh    = (f16*)  alloc((size_t)B_ * T_ * C_ * 2);
  f16*   wqkv  = (f16*)  alloc((size_t)C_ * 3 * C_ * 2);
  float* bqkv  = (float*)alloc((size_t)3 * C_ * 4);
  f16*   wof   = (f16*)  alloc((size_t)C_ * C_ * 2);
  float* qkv   = (float*)alloc((size_t)B_ * T_ * 3 * C_ * 4);
  f16*   qhd   = (f16*)  alloc((size_t)B_ * H_ * T_ * D_ * 2);
  f16*   khd   = (f16*)  alloc((size_t)B_ * H_ * T_ * D_ * 2);
  f16*   vhd   = (f16*)  alloc((size_t)B_ * H_ * T_ * D_ * 2);
  f16*   attn  = (f16*)  alloc((size_t)B_ * T_ * C_ * 2);

  float* out  = (float*)d_out;
  float* kout = out + (size_t)B_ * T_ * C_;
  float* vout = kout + (size_t)B_ * H_ * T_ * D_;

  k_adaln_ss<<<B_, 256, 0, stream>>>(cond, adaln_w, adaln_b, ss);
  k_adaln_norm<<<B_ * T_, 256, 0, stream>>>(s_frames, ss, xh);
  k_pack<<<1536, 256, 0, stream>>>(wq, bq, wk, bk, wv, bv, wo, wqkv, bqkv, wof);

  dim3 g1(B_ * T_ / 64, 3 * C_ / 64);
  k_gemm<<<g1, 256, 0, stream>>>(xh, wqkv, bqkv, qkv, B_ * T_, 3 * C_, C_);

  k_rope<<<B_ * T_, 256, 0, stream>>>(qkv, frame_idx, seq_idx,
                                      qhd, khd, vhd, kout, vout);

  k_attn<<<B_ * H_ * L_ * QTPF_, 128, 0, stream>>>(qhd, khd, vhd,
                                                   ca_pos, mask, attn);

  dim3 g2(B_ * T_ / 64, C_ / 64);
  k_gemm<<<g2, 256, 0, stream>>>(attn, wof, bo, out, B_ * T_, C_, C_);
}
